// RNN_15281493639886
// MI455X (gfx1250) — matrix operand order repaired and checked by reading, not compiled, not run on hardware
//
#include <hip/hip_runtime.h>
#include <hip/hip_bf16.h>

typedef __bf16 bf16;
typedef __attribute__((ext_vector_type(16))) __bf16 bf16x16;
typedef __attribute__((ext_vector_type(8)))  float   f32x8;
typedef __attribute__((ext_vector_type(4)))  unsigned int v4u;
typedef __attribute__((ext_vector_type(8)))  int v8i;
typedef __attribute__((ext_vector_type(4)))  int v4i;

struct U4x2 { uint4 a, b; };

__device__ __forceinline__ bf16 f2bf(float f) {
  unsigned u = __builtin_bit_cast(unsigned, f);
  u += 0x7FFFu + ((u >> 16) & 1u);            // round-to-nearest-even
  unsigned short h = (unsigned short)(u >> 16);
  return __builtin_bit_cast(bf16, h);
}
__device__ __forceinline__ float sigm(float x) { return 1.0f / (1.0f + __expf(-x)); }

// ---------------------------------------------------------------------------
// TDM: async 2-D tile load (rows x 32 bf16 cols) global -> LDS, HW row padding
// D# packing per CDNA5 ISA.
// pad_interval=3 (16 DWORDs = 64B) , pad_amount=3 (4 DWORDs = 16B)
// => LDS row stride 80B = (BK+8) bf16, matching the fragment-read layout.
// ---------------------------------------------------------------------------
__device__ __forceinline__ void tdm_load_tile(const bf16* gptr, int ld_elems,
                                              const bf16* lptr) {
  unsigned long long ga = (unsigned long long)(uintptr_t)gptr;
  unsigned lds = (unsigned)(uintptr_t)lptr;      // flat low 32 bits = LDS offset
  v4u g0 = { 1u,                                  // count=1 (valid), no gather
             lds,                                 // lds_addr
             (unsigned)ga,                        // global_addr[31:0]
             (unsigned)((ga >> 32) & 0x1FFFFFFu) | (2u << 30) }; // [56:32] | type=2
  v8i g1 = { (int)((1u << 16) |                  // data_size = 2 bytes
                   (1u << 20) |                  // pad_enable
                   (3u << 22) |                  // pad_interval: 16 DWORDs
                   (3u << 25)),                  // pad_amount:   4 DWORDs
             (int)(32u << 16),                   // tensor_dim0 = 32  (bits 79:48)
             (int)(128u << 16),                  // tensor_dim1 = 128 (bits 111:80)
             (int)(32u << 16),                   // tile_dim0 = 32    (bits 127:112)
             (int)128,                           // tile_dim1 = 128, tile_dim2 = 0
             (int)ld_elems,                      // tensor_dim0_stride[31:0]
             0, 0 };                             // stride hi, dim1_stride = 0
  v4i gz = { 0, 0, 0, 0 };                       // groups 2/3: dims 2..4 unused
  v8i gz8 = { 0, 0, 0, 0, 0, 0, 0, 0 };          // trailing group (unused/NULL)
  __builtin_amdgcn_tensor_load_to_lds(g0, g1, gz, gz, gz8, 0);
}

// ---------------------------------------------------------------------------
// Utility / prep kernels
// ---------------------------------------------------------------------------
__global__ void k_zero_bf16(bf16* p, long long n) {
  long long i = (long long)blockIdx.x * 256 + threadIdx.x;
  if (i < n) p[i] = __builtin_bit_cast(bf16, (unsigned short)0);
}

// Wcat0 (1024 x 320): [Wih0 (59, pad to 64) | Whh0 (256)], bf16
__global__ void k_pack_wcat0(const float* Wih, const float* Whh, bf16* out) {
  int i = blockIdx.x * 256 + threadIdx.x;
  if (i >= 1024 * 320) return;
  int n = i / 320, k = i % 320;
  float v = 0.f;
  if (k < 59)       v = Wih[n * 59 + k];
  else if (k >= 64) v = Whh[n * 256 + (k - 64)];
  out[i] = f2bf(v);
}

// Wcat1 (1024 x 512): [Wih1 (256) | Whh1 (256)], bf16
__global__ void k_pack_wcat1(const float* Wih, const float* Whh, bf16* out) {
  int i = blockIdx.x * 256 + threadIdx.x;
  if (i >= 1024 * 512) return;
  int n = i >> 9, k = i & 511;
  float v = (k < 256) ? Wih[n * 256 + k] : Whh[n * 256 + (k - 256)];
  out[i] = f2bf(v);
}

__global__ void k_bias_sum(const float* a, const float* b, float* o, int n) {
  int i = blockIdx.x * 256 + threadIdx.x;
  if (i < n) o[i] = a[i] + b[i];
}

// MLP head collapse: tmp = W1@W2 (256x1024)
__global__ void k_tmpw(const float* W1, const float* W2, float* tmp) {
  int i = blockIdx.x * 256 + threadIdx.x;
  if (i >= 256 * 1024) return;
  int r = i >> 10, c = i & 1023;
  float s = 0.f;
  for (int k = 0; k < 512; ++k) s += W1[r * 512 + k] * W2[k * 1024 + c];
  tmp[i] = s;
}
// WcT (512 x 256) bf16 = (tmp @ W3)^T
__global__ void k_wct(const float* tmp, const float* W3, bf16* WcT) {
  int i = blockIdx.x * 256 + threadIdx.x;
  if (i >= 256 * 512) return;
  int r = i >> 9, c = i & 511;
  float s = 0.f;
  for (int k = 0; k < 1024; ++k) s += tmp[r * 1024 + k] * W3[k * 512 + c];
  WcT[c * 256 + r] = f2bf(s);
}
__global__ void k_btmp(const float* b1, const float* W2, const float* b2, float* btmp) {
  int j = blockIdx.x * 256 + threadIdx.x;
  if (j >= 1024) return;
  float s = b2[j];
  for (int k = 0; k < 512; ++k) s += b1[k] * W2[k * 1024 + j];
  btmp[j] = s;
}
__global__ void k_bc(const float* btmp, const float* W3, const float* b3, float* bc) {
  int j = blockIdx.x * 256 + threadIdx.x;
  if (j >= 512) return;
  float s = b3[j];
  for (int k = 0; k < 1024; ++k) s += btmp[k] * W3[k * 512 + j];
  bc[j] = s;
}

// ---------------------------------------------------------------------------
// Feature pipeline
// ---------------------------------------------------------------------------
// nv (B,10,40) f32: [rp@W_rp+b_rp (32) | coords@W_co+b_co (8)]
__global__ void k_nv(const int* ridx, const int* pidx, const float* coords,
                     const float* rtbl, const float* ptbl,
                     const float* Wrp, const float* brp,
                     const float* Wco, const float* bco, float* nv) {
  long long i = (long long)blockIdx.x * 256 + threadIdx.x;
  if (i >= 16384LL * 10 * 40) return;
  int c = (int)(i % 40);
  long long bt = i / 40;
  float s;
  if (c < 32) {
    s = brp[c];
    int r = ridx[bt], p = pidx[bt];
    for (int k = 0; k < 8; ++k) s += rtbl[(long long)r * 8 + k] * Wrp[k * 32 + c];
    for (int k = 0; k < 4; ++k) s += ptbl[(long long)p * 4 + k] * Wrp[(8 + k) * 32 + c];
  } else {
    int cc = c - 32;
    s = bco[cc] + coords[bt * 2 + 0] * Wco[cc] + coords[bt * 2 + 1] * Wco[8 + cc];
  }
  nv[i] = s;
}

// X0 (B,8,64) bf16: [elu(conv) (32) | tanh(emb) (27) | zero pad (5)]
__global__ void k_x0(const float* nv, const float* convw, const float* convb,
                     const int* caridx, const int* weekidx, const int* timeidx,
                     const float* cartbl, const float* weektbl, const float* timetbl,
                     bf16* X0) {
  long long i = (long long)blockIdx.x * 256 + threadIdx.x;
  if (i >= 16384LL * 8 * 64) return;
  int c = (int)(i & 63);
  long long bt = i >> 6;
  int t = (int)(bt & 7);
  long long b = bt >> 3;
  float v = 0.f;
  if (c < 32) {
    v = convb[c];
    const float* nvb = nv + (b * 10 + t) * 40;
    for (int kw = 0; kw < 3; ++kw)
      for (int ic = 0; ic < 40; ++ic)
        v += nvb[kw * 40 + ic] * convw[(kw * 40 + ic) * 32 + c];
    v = v > 0.f ? v : (__expf(v) - 1.f);          // ELU (alpha=1)
  } else if (c < 59) {
    int j = c - 32;
    float e;
    if (j < 16)      e = cartbl[(long long)caridx[b] * 16 + j];
    else if (j < 19) e = weektbl[weekidx[b] * 3 + (j - 16)];
    else             e = timetbl[timeidx[b] * 8 + (j - 19)];
    v = tanhf(e);
  }
  X0[i] = f2bf(v);
}

// ---------------------------------------------------------------------------
// LSTM gate nonlinearity: g (B,1024) -> c (B,256) f32, h -> bf16
// ---------------------------------------------------------------------------
__global__ void k_gate(const float* __restrict__ g, float* __restrict__ c,
                       bf16* __restrict__ hout, int hstride, int first) {
  long long i = (long long)blockIdx.x * 256 + threadIdx.x;
  if (i >= 16384LL * 256) return;
  long long b = i >> 8;
  int n = (int)(i & 255);
  const float* gb = g + b * 1024 + n;
  float gi = gb[0], gf = gb[256], gg = gb[512], go = gb[768];
  float cprev = first ? 0.f : c[i];
  float cn = sigm(gf) * cprev + sigm(gi) * tanhf(gg);
  float h  = sigm(go) * tanhf(cn);
  c[i] = cn;
  hout[b * (long long)hstride + n] = f2bf(h);
}

// ---------------------------------------------------------------------------
// WMMA bf16 GEMM:  C[M,N] = [A0 | A1] (M x (K0+K1), bf16) @ W^T (W is N x K) + bias
// 256 threads = 8 waves, block tile 128x128, K-tile 32.
// Staging via TDM (tensor_load_to_lds) issued by wave 0, double-buffered LDS;
// compute: 4 A-frags x 2 B-frags -> 8 v_wmma_f32_16x16x32_bf16 per K-step.
// ---------------------------------------------------------------------------
#define BM 128
#define BN 128
#define BK 32

__global__ __launch_bounds__(256) void k_gemm(
    const bf16* __restrict__ A0, int lda0, int K0,
    const bf16* __restrict__ A1, int lda1,
    const bf16* __restrict__ W, int ldw, int Ktot,
    const float* __restrict__ bias, float* __restrict__ C, int N) {
  __shared__ __align__(16) bf16 As[2][BM][BK + 8];
  __shared__ __align__(16) bf16 Ws[2][BN][BK + 8];

  const int tid  = threadIdx.x;
  const int lane = tid & 31;
  const int wave = tid >> 5;
  const int wm   = wave & 1;        // 0..1  -> 64 rows each
  const int wn   = wave >> 1;       // 0..3  -> 32 cols each
  const int l16  = lane & 15;
  const int hi   = (lane >> 4) & 1;

  const long long m0 = (long long)blockIdx.y * BM;
  const int n0 = blockIdx.x * BN;

  f32x8 acc[4][2];
#pragma unroll
  for (int i = 0; i < 4; ++i)
#pragma unroll
    for (int j = 0; j < 2; ++j) acc[i][j] = {};

  const int nt = Ktot / BK;

  // Issue both tile DMAs (A and W) for K-tile kt into LDS buffer `buf`.
  auto issue = [&](int kt, int buf) {
    int kk = kt * BK;
    const bf16* Ap; int ld;
    if (kk < K0) { Ap = A0 + kk; ld = lda0; }
    else         { Ap = A1 + (kk - K0); ld = lda1; }
    tdm_load_tile(Ap + m0 * (long long)ld, ld, &As[buf][0][0]);
    tdm_load_tile(W + (long long)n0 * ldw + kk, ldw, &Ws[buf][0][0]);
  };

  if (wave == 0) {                       // TDM ignores EXEC; branch is wave-uniform
    issue(0, 0);
    __builtin_amdgcn_s_wait_tensorcnt(0);
  }
  __syncthreads();

  for (int kt = 0; kt < nt; ++kt) {
    int buf = kt & 1;
    if (kt + 1 < nt && wave == 0) issue(kt + 1, (kt + 1) & 1);  // async prefetch

    // B fragments: lane = column n (l16), lanes<16 hold K 0..15, lanes>=16 K 16..31
    bf16x16 bfr[2];
#pragma unroll
    for (int j = 0; j < 2; ++j) {
      const bf16* wp = &Ws[buf][wn * 32 + j * 16 + l16][hi * 16];
      U4x2 bits;
      bits.a = *(const uint4*)wp;
      bits.b = *(const uint4*)(wp + 8);
      bfr[j] = __builtin_bit_cast(bf16x16, bits);
    }
    // A fragments: lane = row m (l16); lanes<16 K 0..15, lanes>=16 K 16..31
#pragma unroll
    for (int i = 0; i < 4; ++i) {
      const bf16* ap = &As[buf][wm * 64 + i * 16 + l16][hi * 16];
      U4x2 bits;
      bits.a = *(const uint4*)ap;
      bits.b = *(const uint4*)(ap + 8);
      bf16x16 afr = __builtin_bit_cast(bf16x16, bits);
#pragma unroll
      for (int j = 0; j < 2; ++j)
        acc[i][j] = __builtin_amdgcn_wmma_f32_16x16x32_bf16(
            false, afr, false, bfr[j], (short)0, acc[i][j], false, false);
    }

    if (kt + 1 < nt) {
      if (wave == 0) __builtin_amdgcn_s_wait_tensorcnt(0);  // DMA done for kt+1
      __syncthreads();
    }
  }

  // Epilogue: C/D layout -> VGPR r: lanes 0-15 M=r, lanes 16-31 M=r+8; N=l16
#pragma unroll
  for (int i = 0; i < 4; ++i) {
    long long mrow = m0 + wm * 64 + i * 16 + hi * 8;
#pragma unroll
    for (int j = 0; j < 2; ++j) {
      int ncol = n0 + wn * 32 + j * 16 + l16;
      float bv = bias ? bias[ncol] : 0.f;
      float* cp = C + mrow * N + ncol;
#pragma unroll
      for (int r = 0; r < 8; ++r) cp[(long long)r * N] = acc[i][j][r] + bv;
    }
  }
}

// ---------------------------------------------------------------------------
extern "C" void kernel_launch(void* const* d_in, const int* in_sizes, int n_in,
                              void* d_out, int out_size, void* d_ws, size_t ws_size,
                              hipStream_t stream) {
  (void)in_sizes; (void)n_in; (void)out_size; (void)ws_size;

  const int*   region_idx = (const int*)d_in[0];
  const int*   poi_idx    = (const int*)d_in[1];
  const int*   car_idx    = (const int*)d_in[2];
  const int*   week_idx   = (const int*)d_in[3];
  const int*   time_idx   = (const int*)d_in[4];
  const float* coords     = (const float*)d_in[5];
  const float* region_tbl = (const float*)d_in[6];
  const float* poi_tbl    = (const float*)d_in[7];
  const float* car_tbl    = (const float*)d_in[8];
  const float* week_tbl   = (const float*)d_in[9];
  const float* time_tbl   = (const float*)d_in[10];
  const float* W_rp  = (const float*)d_in[11];
  const float* b_rp  = (const float*)d_in[12];
  const float* W_co  = (const float*)d_in[13];
  const float* b_co  = (const float*)d_in[14];
  const float* convw = (const float*)d_in[15];
  const float* convb = (const float*)d_in[16];
  const float* Wih0  = (const float*)d_in[17];
  const float* Whh0  = (const float*)d_in[18];
  const float* bih0  = (const float*)d_in[19];
  const float* bhh0  = (const float*)d_in[20];
  const float* Wih1  = (const float*)d_in[21];
  const float* Whh1  = (const float*)d_in[22];
  const float* bih1  = (const float*)d_in[23];
  const float* bhh1  = (const float*)d_in[24];
  const float* W1    = (const float*)d_in[25];
  const float* b1    = (const float*)d_in[26];
  const float* W2    = (const float*)d_in[27];
  const float* b2    = (const float*)d_in[28];
  const float* W3    = (const float*)d_in[29];
  const float* b3    = (const float*)d_in[30];

  // Workspace carving (all 256B aligned)
  char* wsb = (char*)d_ws;
  auto carve = [&](size_t bytes) {
    char* p = wsb;
    wsb += (bytes + 255) & ~(size_t)255;
    return p;
  };
  float* nv    = (float*)carve(16384ULL * 10 * 40 * 4);   // 26.2 MB
  bf16*  X0    = (bf16*) carve(16384ULL * 8 * 64 * 2);    // 16.8 MB
  bf16*  hs0   = (bf16*) carve(16384ULL * 8 * 256 * 2);   // 67.1 MB
  bf16*  h1    = (bf16*) carve(16384ULL * 256 * 2);       //  8.4 MB
  bf16*  hz    = (bf16*) carve(16384ULL * 256 * 2);       //  8.4 MB (zeros)
  float* c0    = (float*)carve(16384ULL * 256 * 4);       // 16.8 MB
  float* c1    = (float*)carve(16384ULL * 256 * 4);       // 16.8 MB
  float* g     = (float*)carve(16384ULL * 1024 * 4);      // 67.1 MB
  bf16*  Wcat0 = (bf16*) carve(1024ULL * 320 * 2);
  bf16*  Wcat1 = (bf16*) carve(1024ULL * 512 * 2);
  bf16*  WcT   = (bf16*) carve(512ULL * 256 * 2);
  float* tmpW  = (float*)carve(256ULL * 1024 * 4);
  float* bias0 = (float*)carve(1024 * 4);
  float* bias1 = (float*)carve(1024 * 4);
  float* btmp  = (float*)carve(1024 * 4);
  float* bc    = (float*)carve(512 * 4);

  dim3 blk(256);

  // Prep: zero h_init, pack bf16 weights, fold biases, collapse MLP head
  k_zero_bf16<<<16384, blk, 0, stream>>>(hz, 16384LL * 256);
  k_pack_wcat0<<<1280, blk, 0, stream>>>(Wih0, Whh0, Wcat0);
  k_pack_wcat1<<<2048, blk, 0, stream>>>(Wih1, Whh1, Wcat1);
  k_bias_sum<<<4, blk, 0, stream>>>(bih0, bhh0, bias0, 1024);
  k_bias_sum<<<4, blk, 0, stream>>>(bih1, bhh1, bias1, 1024);
  k_tmpw<<<1024, blk, 0, stream>>>(W1, W2, tmpW);
  k_wct<<<512, blk, 0, stream>>>(tmpW, W3, WcT);
  k_btmp<<<4, blk, 0, stream>>>(b1, W2, b2, btmp);
  k_bc<<<2, blk, 0, stream>>>(btmp, W3, b3, bc);

  // Features
  k_nv<<<25600, blk, 0, stream>>>(region_idx, poi_idx, coords, region_tbl, poi_tbl,
                                  W_rp, b_rp, W_co, b_co, nv);
  k_x0<<<32768, blk, 0, stream>>>(nv, convw, convb, car_idx, week_idx, time_idx,
                                  car_tbl, week_tbl, time_tbl, X0);

  // LSTM layer 0: g = [x_t | h_{t-1}] @ Wcat0^T + bias0 (one WMMA GEMM per step)
  for (int t = 0; t < 8; ++t) {
    const bf16* A1 = t ? (hs0 + (long long)(t - 1) * 256) : hz;
    int lda1 = t ? 2048 : 256;
    k_gemm<<<dim3(8, 128), blk, 0, stream>>>(X0 + t * 64, 512, 64, A1, lda1,
                                             Wcat0, 320, 320, bias0, g, 1024);
    k_gate<<<16384, blk, 0, stream>>>(g, c0, hs0 + (long long)t * 256, 2048, t == 0);
  }

  // LSTM layer 1: g = [h0_t | h1_{t-1}] @ Wcat1^T + bias1
  for (int t = 0; t < 8; ++t) {
    const bf16* A1 = t ? h1 : hz;
    k_gemm<<<dim3(8, 128), blk, 0, stream>>>(hs0 + (long long)t * 256, 2048, 256,
                                             A1, 256, Wcat1, 512, 512, bias1, g, 1024);
    k_gate<<<16384, blk, 0, stream>>>(g, c1, h1, 256, t == 0);
  }

  // Collapsed MLP head: out = h1_last @ (W1 W2 W3) + bc
  k_gemm<<<dim3(4, 128), blk, 0, stream>>>(h1, 256, 256, h1, 256,
                                           WcT, 256, 256, bc, (float*)d_out, 512);
}
